// CovDiFF_full_40312563040970
// MI455X (gfx1250) — matrix operand order repaired and checked by reading, not compiled, not run on hardware
//
#include <hip/hip_runtime.h>

#define H_  480
#define W_  640
#define HW  307200
#define C_  128
#define NP  22432
#define IMGSZ ((size_t)C_ * HW)

typedef __attribute__((ext_vector_type(16))) __bf16 bf16x16;
typedef __attribute__((ext_vector_type(8)))  float  f32x8;

union FragB { bf16x16 v; uint4 q[2]; };
union Pack8 { uint4 q; __bf16 h[8]; };

__device__ inline f32x8 zero8() { f32x8 z = {0.f,0.f,0.f,0.f,0.f,0.f,0.f,0.f}; return z; }

__device__ inline f32x8 wmma_bf16(const FragB& a, const FragB& b, f32x8 c) {
  return __builtin_amdgcn_wmma_f32_16x16x32_bf16(false, a.v, false, b.v, (short)0, c, false, false);
}

// -------- Tensor Data Mover availability ------------------------------------
#if defined(__has_builtin)
#  if __has_builtin(__builtin_amdgcn_tensor_load_to_lds)
#    define HAVE_TDM 1
#  endif
#endif
#ifndef HAVE_TDM
#  define HAVE_TDM 0
#endif

#if HAVE_TDM
typedef __attribute__((ext_vector_type(4))) unsigned int u32x4;
typedef __attribute__((ext_vector_type(8))) int i32x8;
typedef __attribute__((ext_vector_type(4))) int i32x4;

// DMA a [rows x 128] bf16 tile (row stride = stride_elems in memory) into LDS at
// lds_off, padding LDS by 16B after each 256B row chunk (row stride 272B in LDS).
// OOB columns (>= rem_elems) read zero / are dropped per TDM spec.
__device__ inline void tdm_load_tile(const __bf16* gsrc, unsigned lds_off,
                                     unsigned rem_elems, unsigned rows,
                                     unsigned stride_elems) {
  unsigned long long ga = (unsigned long long)(uintptr_t)gsrc;
  u32x4 g0;
  g0[0] = 1u;                                         // count=1, user descriptor
  g0[1] = lds_off;                                    // lds_addr (bytes)
  g0[2] = (unsigned)(ga & 0xffffffffu);               // global_addr[31:0]
  g0[3] = (unsigned)((ga >> 32) & 0x01ffffffu) | (2u << 30);  // addr[56:32] | type=2
  i32x8 g1;
  // data_size=2B (code 1), pad_enable, pad_interval=64 DW (code 5), pad_amount=4 DW (code 3)
  g1[0] = (int)((1u << 16) | (1u << 20) | (5u << 22) | (3u << 25));
  g1[1] = (int)((rem_elems & 0xffffu) << 16);                         // tensor_dim0[15:0]
  g1[2] = (int)(((rem_elems >> 16) & 0xffffu) | ((rows & 0xffffu) << 16)); // dim0 hi | dim1 lo
  g1[3] = (int)(((rows >> 16) & 0xffffu) | (128u << 16));             // dim1 hi | tile_dim0=128
  g1[4] = (int)(rows & 0xffffu);                                      // tile_dim1 | tile_dim2=0
  g1[5] = (int)stride_elems;                                          // tensor_dim0_stride[31:0]
  g1[6] = 0;
  g1[7] = 0;
  i32x4 g2 = {1, 0, 0, 0};   // tensor_dim2=1 (2D tile)
  i32x4 g3 = {0, 0, 0, 0};
#if __clang_major__ >= 23
  i32x8 g4 = {0, 0, 0, 0, 0, 0, 0, 0};
  __builtin_amdgcn_tensor_load_to_lds(g0, g1, g2, g3, g4, 0);
#else
  __builtin_amdgcn_tensor_load_to_lds(g0, g1, g2, g3, 0);
#endif
}

__device__ inline void tdm_wait0() {
#if __has_builtin(__builtin_amdgcn_s_wait_tensorcnt)
  __builtin_amdgcn_s_wait_tensorcnt(0);
#else
  asm volatile("s_wait_tensorcnt 0x0" ::: "memory");
#endif
}
#endif  // HAVE_TDM

// ---------------------------------------------------------------- prep
__global__ void prep_kernel(const float* __restrict__ conv_w, const float* __restrict__ lin_w,
                            const float* __restrict__ conv_b, const float* __restrict__ g,
                            const float* __restrict__ be, const float* __restrict__ mu,
                            const float* __restrict__ var,
                            __bf16* __restrict__ wbf, __bf16* __restrict__ linbf,
                            float* __restrict__ bnscale, float* __restrict__ bnshift) {
  int i = blockIdx.x * 256 + threadIdx.x;
  if (i < 128 * 128 * 9) {
    int kx = i % 3, ky = (i / 3) % 3, ci = (i / 9) % 128, co = i / (9 * 128);
    int tap = ky * 3 + kx;
    wbf[(tap * 128 + co) * 128 + ci] = (__bf16)conv_w[i];
  }
  if (i < 128 * 128) linbf[i] = (__bf16)lin_w[i];
  if (i < 128) {
    float s = g[i] * rsqrtf(var[i] + 1e-5f);
    bnscale[i] = s;
    bnshift[i] = (conv_b[i] - mu[i]) * s + be[i];
  }
}

__global__ void img_chlast_kernel(const float* __restrict__ img, __bf16* __restrict__ T) {
  long i = (long)blockIdx.x * 256 + threadIdx.x;   // index = p*128 + c
  if (i >= (long)HW * 128) return;
  int c = (int)(i & 127);
  long p = i >> 7;
  T[i] = (__bf16)img[(long)c * HW + p];
}

__global__ void pcd_bf16_kernel(const float* __restrict__ pcd, __bf16* __restrict__ P) {
  long i = (long)blockIdx.x * 256 + threadIdx.x;
  if (i < (long)NP * 128) P[i] = (__bf16)pcd[i];
}

__global__ void zero_kernel(float* __restrict__ p, int n) {
  int i = blockIdx.x * 256 + threadIdx.x;
  if (i < n) p[i] = 0.f;
}

// ---------------------------------------------------------------- conv (implicit GEMM, WMMA bf16)
__global__ __launch_bounds__(256) void conv_wmma_kernel(
    const __bf16* __restrict__ T,       // [y][x][c]
    const __bf16* __restrict__ wbf,     // [tap][co][ci]
    const float* __restrict__ bnscale, const float* __restrict__ bnshift,
    __bf16* __restrict__ Zc,            // [c][p]
    __bf16* __restrict__ Zp)            // [p][c]
{
  const int lane = threadIdx.x & 31;
  const int wv   = threadIdx.x >> 5;
  const int ln16 = lane & 15;
  const int half = lane >> 4;
  const int x0 = blockIdx.x * 64;
  const int y  = blockIdx.y;
  const int mco = wv * 16 + ln16;

  f32x8 acc[4];
#pragma unroll
  for (int j = 0; j < 4; j++) acc[j] = zero8();

  for (int tap = 0; tap < 9; ++tap) {
    const int dy = tap / 3 - 1, dx = tap % 3 - 1;
    const int yy = y + dy;
    const bool yok = (yy >= 0) && (yy < H_);
#pragma unroll
    for (int cb = 0; cb < 4; ++cb) {
      FragB a;
      {
        const __bf16* wp = wbf + ((size_t)(tap * 128 + mco)) * 128 + cb * 32 + half * 16;
        a.q[0] = *(const uint4*)wp;
        a.q[1] = *(const uint4*)(wp + 8);
      }
#pragma unroll
      for (int j = 0; j < 4; j++) {
        FragB b;
        const int x = x0 + j * 16 + ln16 + dx;
        if (yok && x >= 0 && x < W_) {
          const __bf16* ip = T + ((long)yy * W_ + x) * 128 + cb * 32 + half * 16;
          __builtin_prefetch(ip + 128, 0, 1);
          b.q[0] = *(const uint4*)ip;
          b.q[1] = *(const uint4*)(ip + 8);
        } else {
          b.q[0] = make_uint4(0u, 0u, 0u, 0u);
          b.q[1] = make_uint4(0u, 0u, 0u, 0u);
        }
        acc[j] = wmma_bf16(a, b, acc[j]);
      }
    }
  }
  // epilogue: BN + ReLU, dual-layout store
#pragma unroll
  for (int j = 0; j < 4; j++) {
    const int px = x0 + j * 16 + ln16;
    const long p = (long)y * W_ + px;
    Pack8 pk;
#pragma unroll
    for (int r = 0; r < 8; r++) {
      int c = wv * 16 + half * 8 + r;
      float v = acc[j][r] * bnscale[c] + bnshift[c];
      v = v > 0.f ? v : 0.f;
      __bf16 hb = (__bf16)v;
      Zc[(long)c * HW + p] = hb;
      pk.h[r] = hb;
    }
    *(uint4*)(Zp + p * 128 + wv * 16 + half * 8) = pk.q;
  }
}

// ---------------------------------------------------------------- pcd linear (WMMA bf16)
__global__ __launch_bounds__(256) void pcd_linear_kernel(
    const __bf16* __restrict__ P,    // [n][128]
    const __bf16* __restrict__ Wl,   // [co][ci]
    const float* __restrict__ lb,
    __bf16* __restrict__ Zc,         // [co][NP]
    __bf16* __restrict__ Zt)         // [n][co]
{
  const int lane = threadIdx.x & 31, wv = threadIdx.x >> 5;
  const int ln16 = lane & 15, half = lane >> 4;
  const int n0 = blockIdx.x * 64 + (wv & 3) * 16;
  const int cobase = (wv >> 2) * 64;

  f32x8 acc[4];
#pragma unroll
  for (int j = 0; j < 4; j++) acc[j] = zero8();

  const int arow = n0 + ln16;
  const bool aok = arow < NP;
#pragma unroll
  for (int cb = 0; cb < 4; ++cb) {
    FragB a;
    if (aok) {
      const __bf16* ap = P + (long)arow * 128 + cb * 32 + half * 16;
      a.q[0] = *(const uint4*)ap;
      a.q[1] = *(const uint4*)(ap + 8);
    } else {
      a.q[0] = make_uint4(0u, 0u, 0u, 0u);
      a.q[1] = make_uint4(0u, 0u, 0u, 0u);
    }
#pragma unroll
    for (int j = 0; j < 4; j++) {
      FragB b;
      const int co = cobase + j * 16 + ln16;
      const __bf16* bp = Wl + (long)co * 128 + cb * 32 + half * 16;
      b.q[0] = *(const uint4*)bp;
      b.q[1] = *(const uint4*)(bp + 8);
      acc[j] = wmma_bf16(a, b, acc[j]);
    }
  }
#pragma unroll
  for (int j = 0; j < 4; j++) {
    const int co = cobase + j * 16 + ln16;
    const float bias = lb[co];
    Pack8 pk;
#pragma unroll
    for (int r = 0; r < 8; r++) {
      int n = n0 + half * 8 + r;
      float v = acc[j][r] + bias;
      v = v > 0.f ? v : 0.f;
      __bf16 hb = (__bf16)v;
      pk.h[r] = hb;
      if (n < NP) Zt[(long)n * 128 + co] = hb;
    }
    if (n0 + half * 8 < NP)
      *(uint4*)(Zc + (long)co * NP + n0 + half * 8) = pk.q;  // NP % 8 == 0
  }
}

// ---------------------------------------------------------------- gram = X @ X^T  (WMMA bf16)
// TDM path: DMA [128 x 128] bf16 tiles into LDS (row stride 272B via TDM pad) and
// feed fragments from LDS; fallback path reads fragments straight from global (L2-resident).
#define COV_RS 136  // LDS row stride in elements: 128 + 8 pad (16B) -> bank-conflict free

__global__ __launch_bounds__(256) void cov_gram_kernel(
    const __bf16* __restrict__ X,   // [128][Ktotal]
    float* __restrict__ gram,       // [128][128]
    int Ktotal, int kchunk)
{
  const int lane = threadIdx.x & 31, wv = threadIdx.x >> 5;
  const int ln16 = lane & 15, half = lane >> 4;
  long k0 = (long)blockIdx.x * kchunk;
  long k1 = k0 + kchunk; if (k1 > Ktotal) k1 = Ktotal;
  const int m = wv * 16 + ln16;

  f32x8 acc[8];
#pragma unroll
  for (int j = 0; j < 8; j++) acc[j] = zero8();

#if HAVE_TDM
  __shared__ alignas(16) __bf16 tile[128 * COV_RS];
  for (long kt = k0; kt < k1; kt += 128) {
    if (threadIdx.x < 32) {  // wave 0 issues the DMA (per-wave op, EXEC ignored)
      tdm_load_tile(X + kt, (unsigned)(uintptr_t)&tile[0],
                    (unsigned)(Ktotal - kt), 128u, (unsigned)Ktotal);
      tdm_wait0();
    }
    __syncthreads();
    const int kmax = (int)(((k1 - kt) < 128) ? (k1 - kt) : 128);  // multiple of 32
    for (int kbl = 0; kbl < kmax; kbl += 32) {
      FragB a;
      {
        const __bf16* ap = tile + m * COV_RS + kbl + half * 16;
        a.q[0] = *(const uint4*)ap;
        a.q[1] = *(const uint4*)(ap + 8);
      }
#pragma unroll
      for (int j = 0; j < 8; j++) {
        FragB b;
        const int n = j * 16 + ln16;
        const __bf16* bp = tile + n * COV_RS + kbl + half * 16;
        b.q[0] = *(const uint4*)bp;
        b.q[1] = *(const uint4*)(bp + 8);
        acc[j] = wmma_bf16(a, b, acc[j]);
      }
    }
    __syncthreads();
  }
#else
  for (long kb = k0; kb < k1; kb += 32) {
    FragB a;
    {
      const __bf16* ap = X + (long)m * Ktotal + kb + half * 16;
      __builtin_prefetch(ap + 256, 0, 1);
      a.q[0] = *(const uint4*)ap;
      a.q[1] = *(const uint4*)(ap + 8);
    }
#pragma unroll
    for (int j = 0; j < 8; j++) {
      FragB b;
      const int n = j * 16 + ln16;
      const __bf16* bp = X + (long)n * Ktotal + kb + half * 16;
      b.q[0] = *(const uint4*)bp;
      b.q[1] = *(const uint4*)(bp + 8);
      acc[j] = wmma_bf16(a, b, acc[j]);
    }
  }
#endif

#pragma unroll
  for (int j = 0; j < 8; j++) {
    const int col = j * 16 + ln16;
#pragma unroll
    for (int r = 0; r < 8; r++) {
      const int row = wv * 16 + half * 8 + r;
      atomicAdd(&gram[row * 128 + col], acc[j][r]);
    }
  }
}

// ---------------------------------------------------------------- row sums (means)
__global__ void rowsum_kernel(const __bf16* __restrict__ X, float* __restrict__ sums, int Ktotal) {
  __shared__ float red[256];
  const int c = blockIdx.y;
  const int base = blockIdx.x * 8192;
  int end = base + 8192; if (end > Ktotal) end = Ktotal;
  float s = 0.f;
  for (int k = base + threadIdx.x; k < end; k += 256) s += (float)X[(long)c * Ktotal + k];
  red[threadIdx.x] = s;
  __syncthreads();
  for (int st = 128; st > 0; st >>= 1) {
    if ((int)threadIdx.x < st) red[threadIdx.x] += red[threadIdx.x + st];
    __syncthreads();
  }
  if (threadIdx.x == 0) atomicAdd(&sums[c], red[0]);
}

__global__ void finalize_cov_kernel(const float* __restrict__ gram, const float* __restrict__ sums,
                                    float* __restrict__ cov, int N) {
  int i = blockIdx.x * 256 + threadIdx.x;
  if (i < 16384) {
    int r = i >> 7, c = i & 127;
    float mr = sums[r] / (float)N, mc = sums[c] / (float)N;
    cov[i] = (gram[i] - (float)N * mr * mc) / (float)(N - 1);
  }
}

// ---------------------------------------------------------------- 128x128 inverse (in-place Gauss-Jordan, LDS)
__global__ __launch_bounds__(256) void invert128_kernel(const float* __restrict__ A,
                                                        float* __restrict__ inv, float eps) {
  __shared__ float M[128 * 128];
  for (int i = threadIdx.x; i < 16384; i += 256)
    M[i] = A[i] + (((i >> 7) == (i & 127)) ? eps : 0.f);
  __syncthreads();
  for (int k = 0; k < 128; ++k) {
    const float ip = 1.0f / M[k * 128 + k];
    __syncthreads();
    for (int idx = threadIdx.x; idx < 16384; idx += 256) {
      int i = idx >> 7, j = idx & 127;
      if (i != k && j != k) M[idx] -= M[i * 128 + k] * ip * M[k * 128 + j];
    }
    __syncthreads();
    for (int j = threadIdx.x; j < 128; j += 256) {
      if (j != k) {
        M[k * 128 + j] *= ip;
        M[j * 128 + k] *= -ip;
      }
    }
    if (threadIdx.x == 0) M[k * 128 + k] = ip;
    __syncthreads();
  }
  for (int i = threadIdx.x; i < 16384; i += 256) inv[i] = M[i];
}

// ---------------------------------------------------------------- small 128x128 matmul + softmax + M assembly
__global__ void mm128_kernel(const float* __restrict__ A, const float* __restrict__ B,
                             float* __restrict__ Cm, float scale) {
  int i = blockIdx.x, j = threadIdx.x;
  float s = 0.f;
  for (int k = 0; k < 128; ++k) s += A[i * 128 + k] * B[k * 128 + j];
  Cm[i * 128 + j] = s * scale;
}

__global__ void softmax128_kernel(const float* __restrict__ S, float* __restrict__ A) {
  __shared__ float red[128];
  int i = blockIdx.x, j = threadIdx.x;
  float v = S[i * 128 + j];
  red[j] = v;
  __syncthreads();
  for (int st = 64; st > 0; st >>= 1) {
    if (j < st) red[j] = fmaxf(red[j], red[j + st]);
    __syncthreads();
  }
  float mx = red[0];
  __syncthreads();
  float e = __expf(v - mx);
  red[j] = e;
  __syncthreads();
  for (int st = 64; st > 0; st >>= 1) {
    if (j < st) red[j] += red[j + st];
    __syncthreads();
  }
  A[i * 128 + j] = e / red[0];
}

__global__ void makeM_kernel(const float* __restrict__ a, const float* __restrict__ b,
                             __bf16* __restrict__ Mo) {
  int i = blockIdx.x * 256 + threadIdx.x;
  if (i < 16384) Mo[i] = (__bf16)((a[i] + b[i]) * 0.5f * 0.2f);
}

// ---------------------------------------------------------------- apply: img_out = M_img @ Z + 0.8*img  (WMMA bf16)
__global__ __launch_bounds__(256) void apply_img_kernel(
    const __bf16* __restrict__ Mi,   // [c][k] 128x128 (already *0.5*TAU)
    const __bf16* __restrict__ Zp,   // [p][k]
    const float*  __restrict__ img,  // [c][p]
    float* __restrict__ out)         // [c][p]
{
  const int lane = threadIdx.x & 31, wv = threadIdx.x >> 5;
  const int ln16 = lane & 15, half = lane >> 4;
  const long p0 = (long)blockIdx.x * 64;
  const int mco = wv * 16 + ln16;

  f32x8 acc[4];
#pragma unroll
  for (int j = 0; j < 4; j++) acc[j] = zero8();

#pragma unroll
  for (int cb = 0; cb < 4; ++cb) {
    FragB a;
    {
      const __bf16* ap = Mi + (long)mco * 128 + cb * 32 + half * 16;
      a.q[0] = *(const uint4*)ap;
      a.q[1] = *(const uint4*)(ap + 8);
    }
#pragma unroll
    for (int j = 0; j < 4; j++) {
      FragB b;
      const long p = p0 + j * 16 + ln16;
      const __bf16* bp = Zp + p * 128 + cb * 32 + half * 16;
      __builtin_prefetch(bp + 2048, 0, 1);
      b.q[0] = *(const uint4*)bp;
      b.q[1] = *(const uint4*)(bp + 8);
      acc[j] = wmma_bf16(a, b, acc[j]);
    }
  }
#pragma unroll
  for (int j = 0; j < 4; j++) {
    const long p = p0 + j * 16 + ln16;
#pragma unroll
    for (int r = 0; r < 8; r++) {
      const int c = wv * 16 + half * 8 + r;
      const long idx = (long)c * HW + p;
      out[idx] = acc[j][r] + 0.8f * img[idx];
    }
  }
}

// ---------------------------------------------------------------- apply: pcd_out[n][c] = sum_k Mp[c][k]*Zt[n][k] + 0.8*pcd
__global__ __launch_bounds__(256) void apply_pcd_kernel(
    const __bf16* __restrict__ Zt,   // [n][k]
    const __bf16* __restrict__ Mp,   // [c][k]
    const float*  __restrict__ pcd,  // [n][c]
    float* __restrict__ out)         // [n][c]
{
  const int lane = threadIdx.x & 31, wv = threadIdx.x >> 5;
  const int ln16 = lane & 15, half = lane >> 4;
  const int n0 = blockIdx.x * 64 + (wv & 3) * 16;
  const int cobase = (wv >> 2) * 64;

  f32x8 acc[4];
#pragma unroll
  for (int j = 0; j < 4; j++) acc[j] = zero8();

  const int arow = n0 + ln16;
  const bool aok = arow < NP;
#pragma unroll
  for (int cb = 0; cb < 4; ++cb) {
    FragB a;
    if (aok) {
      const __bf16* ap = Zt + (long)arow * 128 + cb * 32 + half * 16;
      a.q[0] = *(const uint4*)ap;
      a.q[1] = *(const uint4*)(ap + 8);
    } else {
      a.q[0] = make_uint4(0u, 0u, 0u, 0u);
      a.q[1] = make_uint4(0u, 0u, 0u, 0u);
    }
#pragma unroll
    for (int j = 0; j < 4; j++) {
      FragB b;
      const int c = cobase + j * 16 + ln16;
      const __bf16* bp = Mp + (long)c * 128 + cb * 32 + half * 16;
      b.q[0] = *(const uint4*)bp;
      b.q[1] = *(const uint4*)(bp + 8);
      acc[j] = wmma_bf16(a, b, acc[j]);
    }
  }
#pragma unroll
  for (int j = 0; j < 4; j++) {
    const int c = cobase + j * 16 + ln16;
#pragma unroll
    for (int r = 0; r < 8; r++) {
      const int n = n0 + half * 8 + r;
      if (n < NP) {
        const long idx = (long)n * 128 + c;
        out[idx] = acc[j][r] + 0.8f * pcd[idx];
      }
    }
  }
}

// ================================================================ host
extern "C" void kernel_launch(void* const* d_in, const int* in_sizes, int n_in,
                              void* d_out, int out_size, void* d_ws, size_t ws_size,
                              hipStream_t stream) {
  (void)in_sizes; (void)n_in; (void)out_size; (void)ws_size;
  const float* img   = (const float*)d_in[0];
  const float* pcd   = (const float*)d_in[1];
  const float* convw = (const float*)d_in[2];
  const float* convb = (const float*)d_in[3];
  const float* bng   = (const float*)d_in[4];
  const float* bnb   = (const float*)d_in[5];
  const float* bnm   = (const float*)d_in[6];
  const float* bnv   = (const float*)d_in[7];
  const float* linw  = (const float*)d_in[8];
  const float* linb  = (const float*)d_in[9];

  char* ws = (char*)d_ws;
  size_t off = 0;
  auto alloc = [&](size_t bytes) { size_t r = off; off = (off + bytes + 255) & ~(size_t)255; return r; };

  __bf16* Timg   = (__bf16*)(ws + alloc((size_t)HW * 128 * 2));   // [y][x][c]
  __bf16* Zc_img = (__bf16*)(ws + alloc((size_t)HW * 128 * 2));   // [c][p]
  __bf16* Zp_img = (__bf16*)(ws + alloc((size_t)HW * 128 * 2));   // [p][c]
  __bf16* Pbf    = (__bf16*)(ws + alloc((size_t)NP * 128 * 2));   // [n][c]
  __bf16* Zc_pcd = (__bf16*)(ws + alloc((size_t)NP * 128 * 2));   // [c][n]
  __bf16* Zt_pcd = (__bf16*)(ws + alloc((size_t)NP * 128 * 2));   // [n][c]
  __bf16* wbf    = (__bf16*)(ws + alloc(9 * 128 * 128 * 2));
  __bf16* linbf  = (__bf16*)(ws + alloc(128 * 128 * 2));
  float*  bnscale = (float*)(ws + alloc(128 * 4));
  float*  bnshift = (float*)(ws + alloc(128 * 4));
  float* accum    = (float*)(ws + alloc((16384 + 16384 + 128 + 128) * 4));
  float* gram_img = accum;
  float* gram_pcd = accum + 16384;
  float* sum_img  = accum + 32768;
  float* sum_pcd  = accum + 32768 + 128;
  float* cov_img  = (float*)(ws + alloc(16384 * 4));
  float* cov_pcd  = (float*)(ws + alloc(16384 * 4));
  float* inv_img  = (float*)(ws + alloc(16384 * 4));
  float* inv_pcd  = (float*)(ws + alloc(16384 * 4));
  float* sc_ip    = (float*)(ws + alloc(16384 * 4));
  float* sc_pi    = (float*)(ws + alloc(16384 * 4));
  float* g_ii     = (float*)(ws + alloc(16384 * 4));
  float* g_pp     = (float*)(ws + alloc(16384 * 4));
  float* at_sc_ip = (float*)(ws + alloc(16384 * 4));
  float* at_sc_pi = (float*)(ws + alloc(16384 * 4));
  float* at_ii    = (float*)(ws + alloc(16384 * 4));
  float* at_pp    = (float*)(ws + alloc(16384 * 4));
  __bf16* M_img   = (__bf16*)(ws + alloc(16384 * 2));
  __bf16* M_pcd   = (__bf16*)(ws + alloc(16384 * 2));

  float* out_img = (float*)d_out;
  float* out_pcd = (float*)d_out + IMGSZ;

  zero_kernel<<<(33024 + 255) / 256, 256, 0, stream>>>(accum, 33024);
  prep_kernel<<<(147456 + 255) / 256, 256, 0, stream>>>(convw, linw, convb, bng, bnb, bnm, bnv,
                                                        wbf, linbf, bnscale, bnshift);
  img_chlast_kernel<<<(int)(((size_t)HW * 128 + 255) / 256), 256, 0, stream>>>(img, Timg);
  pcd_bf16_kernel<<<(int)(((size_t)NP * 128 + 255) / 256), 256, 0, stream>>>(pcd, Pbf);
  conv_wmma_kernel<<<dim3(W_ / 64, H_), 256, 0, stream>>>(Timg, wbf, bnscale, bnshift, Zc_img, Zp_img);
  pcd_linear_kernel<<<(NP + 63) / 64, 256, 0, stream>>>(Pbf, linbf, linb, Zc_pcd, Zt_pcd);
  rowsum_kernel<<<dim3((HW + 8191) / 8192, 128), 256, 0, stream>>>(Zc_img, sum_img, HW);
  rowsum_kernel<<<dim3((NP + 8191) / 8192, 128), 256, 0, stream>>>(Zc_pcd, sum_pcd, NP);
  cov_gram_kernel<<<(HW + 4095) / 4096, 256, 0, stream>>>(Zc_img, gram_img, HW, 4096);
  cov_gram_kernel<<<(NP + 2047) / 2048, 256, 0, stream>>>(Zc_pcd, gram_pcd, NP, 2048);
  finalize_cov_kernel<<<64, 256, 0, stream>>>(gram_img, sum_img, cov_img, HW);
  finalize_cov_kernel<<<64, 256, 0, stream>>>(gram_pcd, sum_pcd, cov_pcd, NP);
  invert128_kernel<<<1, 256, 0, stream>>>(cov_img, inv_img, 1e-6f);
  invert128_kernel<<<1, 256, 0, stream>>>(cov_pcd, inv_pcd, 1e-6f);
  mm128_kernel<<<128, 128, 0, stream>>>(cov_img, inv_pcd, sc_ip, 1.0f / 128.0f);
  mm128_kernel<<<128, 128, 0, stream>>>(cov_pcd, inv_img, sc_pi, 1.0f / 128.0f);
  mm128_kernel<<<128, 128, 0, stream>>>(cov_img, cov_img, g_ii, 1.0f / 128.0f);
  mm128_kernel<<<128, 128, 0, stream>>>(cov_pcd, cov_pcd, g_pp, 1.0f / 128.0f);
  softmax128_kernel<<<128, 128, 0, stream>>>(sc_ip, at_sc_ip);
  softmax128_kernel<<<128, 128, 0, stream>>>(sc_pi, at_sc_pi);
  softmax128_kernel<<<128, 128, 0, stream>>>(g_ii, at_ii);
  softmax128_kernel<<<128, 128, 0, stream>>>(g_pp, at_pp);
  makeM_kernel<<<64, 256, 0, stream>>>(at_sc_pi, at_ii, M_img);
  makeM_kernel<<<64, 256, 0, stream>>>(at_sc_ip, at_pp, M_pcd);
  apply_img_kernel<<<HW / 64, 256, 0, stream>>>(M_img, Zp_img, img, out_img);
  apply_pcd_kernel<<<(NP + 63) / 64, 256, 0, stream>>>(Zt_pcd, M_pcd, pcd, out_pcd);
}
